// Nonhier_Optmatch_13271448945071
// MI455X (gfx1250) — matrix operand order repaired and checked by reading, not compiled, not run on hardware
//
#include <hip/hip_runtime.h>
#include <hip/hip_bf16.h>

typedef unsigned short u16;
typedef unsigned int   u32;

typedef __attribute__((ext_vector_type(16))) __bf16 v16bf;
typedef __attribute__((ext_vector_type(8)))  float  v8f;

// ---------------- constants ----------------
#define NROWS   262144
#define CF      256
#define DMODEL  512
#define EMB     256
#define NSEAS   8192
#define NBLK    256          // sort blocks (1024 members each)
#define BLKSZ   1024

// ---------------- workspace layout (bytes) ----------------
#define OFF_BPACK   ((size_t)0)           // 256 KB  bf16 weights in B-fragment layout
#define OFF_CVEC    ((size_t)262144)      // 1 KB    fused bias
#define OFF_COUNTS  ((size_t)263168)      // 32 KB   per-season counts
#define OFF_OFFS    ((size_t)295936)      // 32 KB   per-season start offsets
#define OFF_T1      ((size_t)328704)      // 512 KB  Wfc@Wo
#define OFF_BHIST   ((size_t)852992)      // 8 MB    per-block season histograms / starts
#define OFF_PERM    ((size_t)9241600)     // 1 MB    member permutation (sorted by season)
#define OFF_XSW     ((size_t)10485760)    // 256 MB  bf16 x, swizzled to A-fragment layout
#define OFF_EMB     ((size_t)278921216)   // 256 MB  member embeddings (sorted order)

// ---------------- helpers ----------------
__device__ __forceinline__ float dot4(float4 a, float4 b) {
    return a.x*b.x + a.y*b.y + a.z*b.z + a.w*b.w;
}
__device__ __forceinline__ u16 f2bf(float f) {
    union { float f; u32 u; } x; x.f = f;
    u32 r = x.u + 0x7FFFu + ((x.u >> 16) & 1u);   // round-to-nearest-even
    return (u16)(r >> 16);
}
__device__ __forceinline__ u32 pack2(float lo, float hi) {
    return (u32)f2bf(lo) | ((u32)f2bf(hi) << 16);
}
__device__ __forceinline__ float lrelu(float x) { return x > 0.0f ? x : 0.01f * x; }

// ================= weight fusion =================
// T1 = Wfc @ Wo   [256 x 512]
__global__ __launch_bounds__(256) void k_t1(const float* __restrict__ Wfc,
                                            const float* __restrict__ Wo,
                                            float* __restrict__ T1) {
    int idx = blockIdx.x * 256 + threadIdx.x;     // 131072
    int e = idx >> 9, j = idx & 511;
    float s = 0.0f;
    for (int k = 0; k < 512; ++k) s += Wfc[e*512 + k] * Wo[k*512 + j];
    T1[idx] = s;
}

// c = T1@bv + Wfc@bo + bfc
__global__ __launch_bounds__(256) void k_cvec(const float* __restrict__ T1,
                                              const float* __restrict__ Wfc,
                                              const float* __restrict__ bv,
                                              const float* __restrict__ bo,
                                              const float* __restrict__ bfc,
                                              float* __restrict__ cvec) {
    int e = threadIdx.x;
    float s = bfc[e];
    for (int j = 0; j < 512; ++j) s += T1[e*512 + j] * bv[j];
    for (int k = 0; k < 512; ++k) s += Wfc[e*512 + k] * bo[k];
    cvec[e] = s;
}

// M = T1 @ Wv  [256 x 512], written directly into WMMA B-fragment layout (bf16).
// B-fragment order used here, same as A (16-bit B 32x16): lane = n + 16*h, element el -> K = el + 16*h.
__global__ __launch_bounds__(256) void k_mb(const float* __restrict__ T1,
                                            const float* __restrict__ Wv,
                                            u16* __restrict__ Bpack) {
    int idx = blockIdx.x * 256 + threadIdx.x;     // 131072
    int e = idx >> 9, i = idx & 511;              // e = output col, i = K
    float s = 0.0f;
    for (int j = 0; j < 512; ++j) s += T1[e*512 + j] * Wv[j*512 + i];
    int ct = e >> 4, n = e & 15;
    int ks = i >> 5, kk = i & 31, h = kk >> 4, el = kk & 15;
    Bpack[(((ct*16 + ks) * 32) + n + 16*h) * 16 + el] = f2bf(s);
}

// ================= deterministic counting sort by season =================
__global__ __launch_bounds__(256) void k_blockhist(const int* __restrict__ season,
                                                   u32* __restrict__ bhist) {
    __shared__ u32 hist[NSEAS];
    int b = blockIdx.x, t = threadIdx.x;
    for (int i = t; i < NSEAS; i += 256) hist[i] = 0;
    __syncthreads();
    for (int k = 0; k < 4; ++k) {
        int s = season[b*BLKSZ + k*256 + t];
        atomicAdd(&hist[s], 1u);                  // LDS atomic: deterministic counts
    }
    __syncthreads();
    for (int i = t; i < NSEAS; i += 256) bhist[(size_t)b*NSEAS + i] = hist[i];
}

// per-season scan over blocks: bhist -> exclusive running starts, counts[s] = total
__global__ __launch_bounds__(256) void k_totals(u32* __restrict__ bhist,
                                                u32* __restrict__ counts) {
    int s = blockIdx.x * 256 + threadIdx.x;       // 8192 threads
    u32 run = 0;
    for (int b = 0; b < NBLK; ++b) {
        u32 v = bhist[(size_t)b*NSEAS + s];
        bhist[(size_t)b*NSEAS + s] = run;
        run += v;
    }
    counts[s] = run;
}

// exclusive prefix over 8192 season counts (single workgroup)
__global__ __launch_bounds__(256) void k_offsets(const u32* __restrict__ counts,
                                                 u32* __restrict__ offs) {
    __shared__ u32 part[256];
    int t = threadIdx.x, base = t * 32;
    u32 sum = 0;
    for (int i = 0; i < 32; ++i) sum += counts[base + i];
    part[t] = sum;
    __syncthreads();
    for (int off = 1; off < 256; off <<= 1) {
        u32 v = (t >= off) ? part[t - off] : 0u;
        __syncthreads();
        part[t] += v;
        __syncthreads();
    }
    u32 run = part[t] - sum;                      // exclusive
    for (int i = 0; i < 32; ++i) { offs[base + i] = run; run += counts[base + i]; }
}

// rank-based stable scatter: perm[pos] = original index (deterministic, no global atomics)
__global__ __launch_bounds__(256) void k_scatter(const int* __restrict__ season,
                                                 const u32* __restrict__ offs,
                                                 const u32* __restrict__ bhist,
                                                 u32* __restrict__ perm) {
    __shared__ u16 sl[BLKSZ];
    int b = blockIdx.x, t = threadIdx.x;
    for (int i = t; i < BLKSZ; i += 256) sl[i] = (u16)season[b*BLKSZ + i];
    __syncthreads();
    for (int k = 0; k < 4; ++k) {
        int i = k*256 + t;
        u16 s = sl[i];
        int rank = 0;
        for (int j = 0; j < i; ++j) rank += (sl[j] == s);
        perm[offs[s] + bhist[(size_t)b*NSEAS + s] + rank] = (u32)(b*BLKSZ + i);
    }
}

// ================= stage 1: gate + build x (bf16, A-fragment swizzled) =================
// A-fragment order used here (16-bit A 16x32): lane = m + 16*h; element e -> K = e + 16*h.
__global__ __launch_bounds__(256) void k1_x(const float* __restrict__ indiv,
                                            const float* __restrict__ H,
                                            const float* __restrict__ S,
                                            const float* __restrict__ R,
                                            const float* __restrict__ w1,
                                            const u32* __restrict__ perm,
                                            u16* __restrict__ xsw) {
    __shared__ u16 xl[16 * 512];                  // 16 rows x 512 bf16
    const int tid = threadIdx.x, wv = tid >> 5, lane = tid & 31;
    const int g = blockIdx.x;                     // 16-row tile id (sorted order)
    const float4* H4 = (const float4*)H;
    const float4* S4 = (const float4*)S;
    const float4* R4 = (const float4*)R;
    const float4* I4 = (const float4*)indiv;
    const float4* W4 = (const float4*)w1;
    float4 wa0 = W4[lane*2], wa1 = W4[lane*2 + 1];
    float4 wb0 = W4[64 + lane*2], wb1 = W4[64 + lane*2 + 1];

    for (int t = 0; t < 2; ++t) {
        int m = wv*2 + t;                         // local row 0..15
        u32 orig = perm[g*16 + m];
        size_t base = (size_t)orig * 64 + lane * 2;
        float4 h0 = H4[base], h1 = H4[base + 1];
        float4 s0 = S4[base], s1 = S4[base + 1];
        float4 r0 = R4[base], r1 = R4[base + 1];
        float pHa = dot4(h0, wa0) + dot4(h1, wa1);
        float pSb = dot4(s0, wb0) + dot4(s1, wb1);
        float pRb = dot4(r0, wb0) + dot4(r1, wb1);
        float pHb = dot4(h0, wb0) + dot4(h1, wb1);
        for (int off = 16; off > 0; off >>= 1) {
            pHa += __shfl_xor(pHa, off);
            pSb += __shfl_xor(pSb, off);
            pRb += __shfl_xor(pRb, off);
            pHb += __shfl_xor(pHb, off);
        }
        float e1 = lrelu(pHa + pSb), e2 = lrelu(pHa + pRb), e3 = lrelu(pHa + pHb);
        float mx = fmaxf(e1, fmaxf(e2, e3));
        float a1 = expf(e1 - mx), a2 = expf(e2 - mx), a3 = expf(e3 - mx);
        float inv = 1.0f / (a1 + a2 + a3);
        a1 *= inv; a2 *= inv; a3 *= inv;

        float4 i0 = I4[base], i1 = I4[base + 1];
        u32* xlw = (u32*)xl;
        int rb = m*256 + lane*4;                  // u32 index (2 bf16 per u32)
        xlw[rb + 0] = pack2(i0.x, i0.y);
        xlw[rb + 1] = pack2(i0.z, i0.w);
        xlw[rb + 2] = pack2(i1.x, i1.y);
        xlw[rb + 3] = pack2(i1.z, i1.w);
        float gx0 = a1*s0.x + a2*r0.x + a3*h0.x;
        float gy0 = a1*s0.y + a2*r0.y + a3*h0.y;
        float gz0 = a1*s0.z + a2*r0.z + a3*h0.z;
        float gw0 = a1*s0.w + a2*r0.w + a3*h0.w;
        float gx1 = a1*s1.x + a2*r1.x + a3*h1.x;
        float gy1 = a1*s1.y + a2*r1.y + a3*h1.y;
        float gz1 = a1*s1.z + a2*r1.z + a3*h1.z;
        float gw1 = a1*s1.w + a2*r1.w + a3*h1.w;
        xlw[rb + 128 + 0] = pack2(gx0, gy0);
        xlw[rb + 128 + 1] = pack2(gz0, gw0);
        xlw[rb + 128 + 2] = pack2(gx1, gy1);
        xlw[rb + 128 + 3] = pack2(gz1, gw1);
    }
    __syncthreads();

    // swizzled writeout: 1024 x 16B chunks per tile, perfectly coalesced
    uint4* dst = (uint4*)xsw + (size_t)g * 1024;
    for (int it = 0; it < 4; ++it) {
        int q = it*256 + tid;
        int ks = q >> 6, rem = q & 63, la = rem >> 1, e0 = (rem & 1) << 3;
        int m = la & 15, hh = la >> 4;
        int K0 = ks*32 + (hh << 4) + e0;
        dst[q] = *(const uint4*)&xl[m*512 + K0];
    }
}

// ================= stage 2: WMMA GEMM  emb = relu(x @ M^T + c) =================
__global__ __launch_bounds__(256) void k2_gemm(const u16* __restrict__ xsw,
                                               const u16* __restrict__ Bpack,
                                               const float* __restrict__ cvec,
                                               float* __restrict__ emb) {
    extern __shared__ u16 Bl[];                   // 256 KB: full fused weight matrix
    const int tid = threadIdx.x;
    const int wv = tid >> 5, lane = tid & 31;
    const size_t rb = (size_t)blockIdx.x * 8 + wv;   // 16-row tile per wave

    // issue first A-fragment load early: overlaps the 256 KB LDS fill
    const v16bf* Abase = (const v16bf*)xsw + rb*16*32 + lane;
    v16bf a = Abase[0];

    {   // cooperative copy global -> LDS (L2 resident after first WG)
        const uint4* src = (const uint4*)Bpack;
        uint4* dst = (uint4*)Bl;
        for (int i = tid; i < 16384; i += 256) dst[i] = src[i];
    }
    __syncthreads();

    const v16bf* Bp = (const v16bf*)Bl;

    v8f acc[16] = {};
    v16bf b = Bp[lane];                           // B[ks=0][ct=0]
    #pragma unroll 1
    for (int ks = 0; ks < 16; ++ks) {
        // software-pipelined A: issue next fragment before this iteration's WMMAs,
        // so its wait lands across the loop backedge behind 16 WMMAs of work
        v16bf an = Abase[(size_t)((ks < 15) ? (ks + 1) : 15) * 32];
        #pragma unroll
        for (int ct = 0; ct < 16; ++ct) {
            // 1-deep B prefetch chain: issue B[ct+1] (or B[ks+1][0] at the seam)
            // before the WMMA that consumes B[ct]
            int ctn = (ct + 1) & 15;
            int ksn = (ct == 15) ? ((ks < 15) ? (ks + 1) : 15) : ks;
            v16bf bn = Bp[((ctn*16 + ksn) << 5) + lane];
            acc[ct] = __builtin_amdgcn_wmma_f32_16x16x32_bf16(
                false, a, false, b, (short)0, acc[ct], false, false);
            b = bn;
        }
        a = an;
    }

    // epilogue: bias + relu + store (C/D layout: vgpr j -> row j + 8*(lane>=16), col lane&15)
    const int r0 = (int)rb * 16;
    const int hsel = lane >> 4, n = lane & 15;
    #pragma unroll
    for (int ct = 0; ct < 16; ++ct) {
        int col = ct*16 + n;
        float cb = cvec[col];
        #pragma unroll
        for (int j = 0; j < 8; ++j) {
            float v = acc[ct][j] + cb;
            v = v > 0.0f ? v : 0.0f;
            emb[(size_t)(r0 + j + 8*hsel) * 256 + col] = v;
        }
    }
}

// ================= stage 3: per-season mean + output dot =================
__global__ __launch_bounds__(256) void k3_out(const float* __restrict__ emb,
                                              const u32* __restrict__ offs,
                                              const u32* __restrict__ counts,
                                              const float* __restrict__ Wout,
                                              const float* __restrict__ bout,
                                              float* __restrict__ out) {
    const int wv = threadIdx.x >> 5, lane = threadIdx.x & 31;
    const int s = blockIdx.x * 8 + wv;            // one wave per season
    const u32 start = offs[s], cnt = counts[s];
    float4 a0 = {0,0,0,0}, a1 = {0,0,0,0};
    const float4* E4 = (const float4*)emb;
    for (u32 r = 0; r < cnt; ++r) {
        size_t rowb = (size_t)(start + r) * 64;
        float4 v0 = E4[rowb + lane*2], v1 = E4[rowb + lane*2 + 1];
        a0.x += v0.x; a0.y += v0.y; a0.z += v0.z; a0.w += v0.w;
        a1.x += v1.x; a1.y += v1.y; a1.z += v1.z; a1.w += v1.w;
    }
    float inv = 1.0f / (float)(cnt ? cnt : 1u);
    const float4* W4 = (const float4*)Wout;
    float4 w0 = W4[lane*2], w1 = W4[lane*2 + 1];
    float p = inv * (dot4(a0, w0) + dot4(a1, w1));
    for (int off = 16; off > 0; off >>= 1) p += __shfl_xor(p, off);
    if (lane == 0) out[s] = p + bout[0];
}

// ================= launch =================
extern "C" void kernel_launch(void* const* d_in, const int* in_sizes, int n_in,
                              void* d_out, int out_size, void* d_ws, size_t ws_size,
                              hipStream_t stream) {
    (void)in_sizes; (void)n_in; (void)out_size; (void)ws_size;

    const float* indiv  = (const float*)d_in[0];
    const float* H      = (const float*)d_in[1];
    const float* S      = (const float*)d_in[2];
    const float* R      = (const float*)d_in[3];
    const int*   season = (const int*)  d_in[4];
    const float* w1     = (const float*)d_in[5];
    // d_in[6..9] = Wq,bq,Wk,bk: dead (softmax over size-1 axis == 1)
    const float* Wv     = (const float*)d_in[10];
    const float* bv     = (const float*)d_in[11];
    const float* Wo     = (const float*)d_in[12];
    const float* bo     = (const float*)d_in[13];
    const float* Wfc    = (const float*)d_in[14];
    const float* bfc    = (const float*)d_in[15];
    const float* Wout   = (const float*)d_in[16];
    const float* bout   = (const float*)d_in[17];
    float* out = (float*)d_out;

    char* w = (char*)d_ws;
    u16*   Bpack  = (u16*)  (w + OFF_BPACK);
    float* cvec   = (float*)(w + OFF_CVEC);
    u32*   counts = (u32*)  (w + OFF_COUNTS);
    u32*   offs   = (u32*)  (w + OFF_OFFS);
    float* T1     = (float*)(w + OFF_T1);
    u32*   bhist  = (u32*)  (w + OFF_BHIST);
    u32*   perm   = (u32*)  (w + OFF_PERM);
    u16*   xsw    = (u16*)  (w + OFF_XSW);
    float* embf   = (float*)(w + OFF_EMB);

    // weight fusion: M = Wfc@Wo@Wv, c = Wfc@(Wo@bv+bo)+bfc
    k_t1  <<<512, 256, 0, stream>>>(Wfc, Wo, T1);
    k_cvec<<<1,   256, 0, stream>>>(T1, Wfc, bv, bo, bfc, cvec);
    k_mb  <<<512, 256, 0, stream>>>(T1, Wv, Bpack);

    // deterministic counting sort of members by season
    k_blockhist<<<NBLK, 256, 0, stream>>>(season, bhist);
    k_totals   <<<NSEAS/256, 256, 0, stream>>>(bhist, counts);
    k_offsets  <<<1, 256, 0, stream>>>(counts, offs);
    k_scatter  <<<NBLK, 256, 0, stream>>>(season, offs, bhist, perm);

    // stage 1: gating + bf16 swizzled x
    k1_x<<<NROWS/16, 256, 0, stream>>>(indiv, H, S, R, w1, perm, xsw);

    // stage 2: WMMA GEMM (256 KB dynamic LDS for the weight matrix)
    hipFuncSetAttribute(reinterpret_cast<const void*>(k2_gemm),
                        hipFuncAttributeMaxDynamicSharedMemorySize, 262144);
    k2_gemm<<<NROWS/128, 256, 262144, stream>>>(xsw, Bpack, cvec, embf);

    // stage 3: per-season mean + output linear
    k3_out<<<NSEAS/8, 256, 0, stream>>>(embf, offs, counts, Wout, bout, out);
}
